// RNN_41807211659578
// MI455X (gfx1250) — matrix operand order repaired and checked by reading, not compiled, not run on hardware
//
#include <hip/hip_runtime.h>

// ---------------------------------------------------------------------------
// CDNA5 (gfx1250) fused GRU-scan kernel. wave32, WMMA f32_16x16x32_f16.
// ---------------------------------------------------------------------------

typedef __attribute__((ext_vector_type(16))) _Float16 v16h;
typedef __attribute__((ext_vector_type(8)))  float    v8f;
typedef _Float16 h16;

#define T_      48
#define B_      256
#define D_      92          // NA*NF = 23*4
#define H_      128
#define KENC_   224         // 2 + 92 + 128 = 222 padded to 224 (K-tile multiple of 32)
#define G3H_    384         // 3*H
#define BURN_   12
#define MROWS0  3072        // 12*256
#define MROWS1  2816        // 11*256
#define BLOCKS0 (MROWS0 / 16)
#define BLOCKS1 (MROWS1 / 16)
#define NTHREADS 256
#define NWAVES   8

// ---- WMMA fragment helpers (layouts per CDNA5 ISA 7.12.2) ------

// A-matrix 16x32 f16: lane L -> row M = L&15.
// lanes 0-15 : vgpr0-3 = K 0..7 (pairs), vgpr4-7 = K 16..23
// lanes 16-31: vgpr0-3 = K 8..15,        vgpr4-7 = K 24..31
__device__ inline v16h load_a_frag(const h16* base, int ldk, int k0, int lane) {
    int row = lane & 15;
    int hi  = lane >> 4;
    const h16* r = base + (size_t)row * ldk;
    union { v16h v; unsigned u[8]; } f;
#pragma unroll
    for (int i = 0; i < 4; ++i) {
        int kA = k0 + hi * 8 + 2 * i;         // K 0..15 half
        int kB = k0 + 16 + hi * 8 + 2 * i;    // K 16..31 half
        f.u[i]     = *(const unsigned*)(r + kA);
        f.u[4 + i] = *(const unsigned*)(r + kB);
    }
    return f.v;
}

// B-matrix 32x16 f16, sourced from W stored row-major [N][K] (B[k][n]=W[n][k]).
// lane L -> col N = L&15; K order matches A: lanes 0-15 K 0..7,16..23; lanes 16-31 K 8..15,24..31.
__device__ inline v16h load_b_frag(const h16* W, int ldk, int n0, int k0, int lane) {
    int n  = n0 + (lane & 15);
    int hi = lane >> 4;
    const h16* r = W + (size_t)n * ldk + k0;
    union { v16h v; unsigned u[8]; } f;
#pragma unroll
    for (int i = 0; i < 4; ++i) {
        f.u[i]     = *(const unsigned*)(r + hi * 8 + 2 * i);
        f.u[4 + i] = *(const unsigned*)(r + 16 + hi * 8 + 2 * i);
    }
    return f.v;
}

__device__ inline v8f wmma16(v16h a, v16h b, v8f c) {
    return __builtin_amdgcn_wmma_f32_16x16x32_f16(
        /*neg_a=*/false, a, /*neg_b=*/false, b,
        /*c_mod=*/(short)0, c, /*reuse_a=*/false, /*reuse_b=*/false);
}

// C/D 16x16 f32: vgpr j -> M = j (lanes 0-15) or 8+j (lanes 16-31); N = lane&15.
// One 16-row M tile, nNt N-tiles distributed over 8 waves.
__device__ inline void gemm_acc(const h16* sA, int ldkA, int k0A, int nKt,
                                const h16* W, int ldkW,
                                float* sOut, int ldOut, int nNt,
                                int wave, int lane) {
    for (int tn = wave; tn < nNt; tn += NWAVES) {
        int n0 = tn * 16;
        v8f acc = {};
        for (int kk = 0; kk < nKt; ++kk) {
            v16h a = load_a_frag(sA, ldkA, k0A + kk * 32, lane);
            v16h b = load_b_frag(W, ldkW, n0, kk * 32, lane);
            acc = wmma16(a, b, acc);
        }
        int n  = n0 + (lane & 15);
        int mb = (lane >> 4) * 8;
#pragma unroll
        for (int j = 0; j < 8; ++j)
            sOut[(size_t)(mb + j) * ldOut + n] = acc[j];
    }
}

__device__ inline float sigmoidf_(float x) { return 1.0f / (1.0f + __expf(-x)); }

// ---- fused per-timestep kernel: both groups, full GRU(2) + MLP + errors ----

__global__ __launch_bounds__(NTHREADS) void step_kernel(
    const float* __restrict__ states, int t,
    const h16* __restrict__ Wih0f, const h16* __restrict__ Whh0f,
    const h16* __restrict__ Wih1f, const h16* __restrict__ Whh1f,
    const h16* __restrict__ W1f,   const h16* __restrict__ W2f,
    const float* __restrict__ b_ih0, const float* __restrict__ b_hh0,
    const float* __restrict__ b_ih1, const float* __restrict__ b_hh1,
    const float* __restrict__ b1,    const float* __restrict__ b2,
    const float* __restrict__ Wm,    const float* __restrict__ bm,
    h16* h0g0, h16* h1g0, h16* h0g1, h16* h1g1,
    float* accum)
{
    __shared__ h16   sX[16 * KENC_];     // enc_in tile: [vel(2) | y_t(92) | h1_prev(128) | pad(2)]
    __shared__ h16   sBufA[16 * H_];     // h0_new, later d1
    __shared__ h16   sBufB[16 * H_];     // d2
    __shared__ float sAccA[16 * G3H_];   // gi accumulators
    __shared__ float sAccB[16 * G3H_];   // gh accumulators

    const int tid  = threadIdx.x;
    const int lane = tid & 31;
    const int wave = tid >> 5;

    int g, blockRow, lo;
    if ((int)blockIdx.x < BLOCKS0) { g = 0; blockRow = blockIdx.x;           lo = 0;  }
    else                           { g = 1; blockRow = blockIdx.x - BLOCKS0; lo = 12; }

    h16* h0 = g ? h0g1 : h0g0;
    h16* h1 = g ? h1g1 : h1g0;
    const int m0 = blockRow * 16;

    const h16* Wih0g = Wih0f + (size_t)g * G3H_ * KENC_;
    const h16* Whh0g = Whh0f + (size_t)g * G3H_ * H_;
    const h16* Wih1g = Wih1f + (size_t)g * G3H_ * H_;
    const h16* Whh1g = Whh1f + (size_t)g * G3H_ * H_;
    const h16* W1g   = W1f   + (size_t)g * H_ * H_;
    const h16* W2g   = W2f   + (size_t)g * H_ * H_;
    const float* bih0g = b_ih0 + g * G3H_;
    const float* bhh0g = b_hh0 + g * G3H_;
    const float* bih1g = b_ih1 + g * G3H_;
    const float* bhh1g = b_hh1 + g * G3H_;
    const float* b1g = b1 + g * H_;
    const float* b2g = b2 + g * H_;
    const float* Wmg = Wm + g * 2 * H_;
    const float* bmg = bm + g * 2;

    // ---- phase 0: build enc_in tile in LDS --------------------------------
    for (int idx = tid; idx < 16 * KENC_; idx += NTHREADS) {
        int ml = idx / KENC_;
        int k  = idx - ml * KENC_;
        int gm = m0 + ml;
        int a  = gm >> 8;           // agent index in group (B=256)
        int b  = gm & 255;          // batch index
        float v;
        const size_t srow = ((size_t)t * B_ + b) * D_;
        if (k < 2)             v = states[srow + (lo + a) * 4 + 2 + k];  // vel
        else if (k < 2 + D_)   v = states[srow + (k - 2)];               // y_rep
        else if (k < 2 + D_ + H_) v = (float)h1[(size_t)gm * H_ + (k - 94)];
        else                   v = 0.0f;                                  // pad
        sX[idx] = (h16)v;
    }
    __syncthreads();

    // ---- phase 1: layer-0 GEMMs  gi0 = enc_in @ Wih0.T, gh0 = h0 @ Whh0.T --
    gemm_acc(sX, KENC_, 0, KENC_ / 32, Wih0g, KENC_, sAccA, G3H_, G3H_ / 16, wave, lane);
    gemm_acc(h0 + (size_t)m0 * H_, H_, 0, H_ / 32, Whh0g, H_, sAccB, G3H_, G3H_ / 16, wave, lane);
    __syncthreads();

    // ---- phase 2: layer-0 gates -> h0_new ---------------------------------
    for (int idx = tid; idx < 16 * H_; idx += NTHREADS) {
        int ml = idx >> 7, j = idx & 127;
        int gm = m0 + ml;
        float gr  = sAccA[ml * G3H_ + j]        + bih0g[j]
                  + sAccB[ml * G3H_ + j]        + bhh0g[j];
        float gz  = sAccA[ml * G3H_ + H_ + j]   + bih0g[H_ + j]
                  + sAccB[ml * G3H_ + H_ + j]   + bhh0g[H_ + j];
        float gin = sAccA[ml * G3H_ + 2*H_ + j] + bih0g[2*H_ + j];
        float ghn = sAccB[ml * G3H_ + 2*H_ + j] + bhh0g[2*H_ + j];
        float r = sigmoidf_(gr);
        float z = sigmoidf_(gz);
        float n = tanhf(gin + r * ghn);
        float hold = (float)h0[(size_t)gm * H_ + j];
        float hn = (1.0f - z) * n + z * hold;
        sBufA[ml * H_ + j] = (h16)hn;
        h0[(size_t)gm * H_ + j] = (h16)hn;
    }
    __syncthreads();

    // ---- phase 3: layer-1 GEMMs  gi1 = h0_new @ Wih1.T, gh1 = h1_prev @ Whh1.T
    gemm_acc(sBufA, H_, 0, H_ / 32, Wih1g, H_, sAccA, G3H_, G3H_ / 16, wave, lane);
    gemm_acc(sX, KENC_, 94, H_ / 32, Whh1g, H_, sAccB, G3H_, G3H_ / 16, wave, lane);
    __syncthreads();

    // ---- phase 4: layer-1 gates -> h1_new (h1_prev kept in sX) ------------
    for (int idx = tid; idx < 16 * H_; idx += NTHREADS) {
        int ml = idx >> 7, j = idx & 127;
        int gm = m0 + ml;
        float gr  = sAccA[ml * G3H_ + j]        + bih1g[j]
                  + sAccB[ml * G3H_ + j]        + bhh1g[j];
        float gz  = sAccA[ml * G3H_ + H_ + j]   + bih1g[H_ + j]
                  + sAccB[ml * G3H_ + H_ + j]   + bhh1g[H_ + j];
        float gin = sAccA[ml * G3H_ + 2*H_ + j] + bih1g[2*H_ + j];
        float ghn = sAccB[ml * G3H_ + 2*H_ + j] + bhh1g[2*H_ + j];
        float r = sigmoidf_(gr);
        float z = sigmoidf_(gz);
        float n = tanhf(gin + r * ghn);
        float hold = (float)sX[ml * KENC_ + 94 + j];   // h1_prev
        float hn = (1.0f - z) * n + z * hold;
        h1[(size_t)gm * H_ + j] = (h16)hn;
    }
    __syncthreads();

    // ---- phase 5: MLP d1 = relu(h1_prev @ W1.T + b1) ----------------------
    gemm_acc(sX, KENC_, 94, H_ / 32, W1g, H_, sAccA, H_, H_ / 16, wave, lane);
    __syncthreads();
    for (int idx = tid; idx < 16 * H_; idx += NTHREADS) {
        int j = idx & 127;
        sBufA[idx] = (h16)fmaxf(0.0f, sAccA[idx] + b1g[j]);
    }
    __syncthreads();

    // ---- phase 6: MLP d2 = relu(d1 @ W2.T + b2) ---------------------------
    gemm_acc(sBufA, H_, 0, H_ / 32, W2g, H_, sAccB, H_, H_ / 16, wave, lane);
    __syncthreads();
    for (int idx = tid; idx < 16 * H_; idx += NTHREADS) {
        int j = idx & 127;
        sBufB[idx] = (h16)fmaxf(0.0f, sAccB[idx] + b2g[j]);
    }
    __syncthreads();

    // ---- phase 7: dm = d2 @ Wm.T + bm, per-row errors, atomic accumulate --
    if (tid < 16) {
        int gm = m0 + tid;
        float dm0 = bmg[0], dm1 = bmg[1];
#pragma unroll 8
        for (int j = 0; j < H_; ++j) {
            float dv = (float)sBufB[tid * H_ + j];
            dm0 += dv * Wmg[j];
            dm1 += dv * Wmg[H_ + j];
        }
        int a = gm >> 8, b = gm & 255;
        const float* s0 = states + ((size_t)t * B_ + b) * D_ + (lo + a) * 4;
        const float* s1 = states + ((size_t)(t + 1) * B_ + b) * D_ + (lo + a) * 4;
        float p0 = s0[0], p1 = s0[1], v0 = s0[2], v1 = s0[3];
        float q0 = s1[0], q1 = s1[1], w0 = s1[2], w1 = s1[3];
        float dL0 = dm0 - w0, dL1 = dm1 - w1;
        float L  = sqrtf(dL0 * dL0 + dL1 * dL1);
        float e0 = p0 + v0 * 0.1f - q0, e1 = p1 + v1 * 0.1f - q1;
        float ep = sqrtf(e0 * e0 + e1 * e1);
        atomicAdd(&accum[0], L);
        if (t >= BURN_) {
            atomicAdd(&accum[1], ep);
            atomicAdd(&accum[2], L);   // ev uses dm[:, :2] == dm -> same as L
        }
    }
}

// ---- prep / epilogue kernels ----------------------------------------------

__global__ void cvt_kernel(const float* __restrict__ src, h16* __restrict__ dst, int n) {
    int i = blockIdx.x * blockDim.x + threadIdx.x;
    if (i < n) dst[i] = (h16)src[i];
}

__global__ void cvt_pad_kernel(const float* __restrict__ src, h16* __restrict__ dst,
                               int ksrc, int kdst, int n) {
    int i = blockIdx.x * blockDim.x + threadIdx.x;
    if (i < n) {
        int r = i / kdst, k = i - r * kdst;
        dst[i] = (k < ksrc) ? (h16)src[r * ksrc + k] : (h16)0.0f;
    }
}

__global__ void zero_h_kernel(h16* p, int n) {
    int i = blockIdx.x * blockDim.x + threadIdx.x;
    if (i < n) p[i] = (h16)0.0f;
}

__global__ void zero_acc_kernel(float* a) {
    if (threadIdx.x < 3) a[threadIdx.x] = 0.0f;
}

__global__ void final_kernel(const float* __restrict__ accum, float* __restrict__ out) {
    if (threadIdx.x == 0) {
        out[0] = accum[0] / (float)(T_ * 23);
        out[1] = accum[1] / (float)((T_ - BURN_) * 23);
        out[2] = accum[2] / (float)((T_ - BURN_) * 23);
    }
}

// ---------------------------------------------------------------------------

extern "C" void kernel_launch(void* const* d_in, const int* in_sizes, int n_in,
                              void* d_out, int out_size, void* d_ws, size_t ws_size,
                              hipStream_t stream) {
    const float* states = (const float*)d_in[0];
    const float* W_ih0  = (const float*)d_in[1];
    const float* W_hh0  = (const float*)d_in[2];
    const float* b_ih0  = (const float*)d_in[3];
    const float* b_hh0  = (const float*)d_in[4];
    const float* W_ih1  = (const float*)d_in[5];
    const float* W_hh1  = (const float*)d_in[6];
    const float* b_ih1  = (const float*)d_in[7];
    const float* b_hh1  = (const float*)d_in[8];
    const float* W1     = (const float*)d_in[9];
    const float* b1     = (const float*)d_in[10];
    const float* W2     = (const float*)d_in[11];
    const float* b2     = (const float*)d_in[12];
    const float* Wm     = (const float*)d_in[13];
    const float* bm     = (const float*)d_in[14];

    // workspace layout (256B-aligned blocks)
    char* base = (char*)d_ws;
    size_t off = 0;
    auto walloc = [&](size_t bytes) -> void* {
        void* p = base + off;
        off += (bytes + 255) & ~(size_t)255;
        return p;
    };
    float* accum = (float*)walloc(3 * sizeof(float));
    h16* h0g0 = (h16*)walloc((size_t)MROWS0 * H_ * sizeof(h16));
    h16* h1g0 = (h16*)walloc((size_t)MROWS0 * H_ * sizeof(h16));
    h16* h0g1 = (h16*)walloc((size_t)MROWS1 * H_ * sizeof(h16));
    h16* h1g1 = (h16*)walloc((size_t)MROWS1 * H_ * sizeof(h16));
    h16* Wih0f = (h16*)walloc((size_t)2 * G3H_ * KENC_ * sizeof(h16));
    h16* Whh0f = (h16*)walloc((size_t)2 * G3H_ * H_ * sizeof(h16));
    h16* Wih1f = (h16*)walloc((size_t)2 * G3H_ * H_ * sizeof(h16));
    h16* Whh1f = (h16*)walloc((size_t)2 * G3H_ * H_ * sizeof(h16));
    h16* W1f   = (h16*)walloc((size_t)2 * H_ * H_ * sizeof(h16));
    h16* W2f   = (h16*)walloc((size_t)2 * H_ * H_ * sizeof(h16));
    (void)ws_size; (void)in_sizes; (void)n_in; (void)out_size;

    auto blocks = [](int n) { return (n + 255) / 256; };

    // zero persistent state + accumulators (ws is not re-poisoned by harness,
    // but we must not rely on leftover state either)
    zero_acc_kernel<<<1, 32, 0, stream>>>(accum);
    zero_h_kernel<<<blocks(MROWS0 * H_), 256, 0, stream>>>(h0g0, MROWS0 * H_);
    zero_h_kernel<<<blocks(MROWS0 * H_), 256, 0, stream>>>(h1g0, MROWS0 * H_);
    zero_h_kernel<<<blocks(MROWS1 * H_), 256, 0, stream>>>(h0g1, MROWS1 * H_);
    zero_h_kernel<<<blocks(MROWS1 * H_), 256, 0, stream>>>(h1g1, MROWS1 * H_);

    // convert weights to f16 (W_ih0 padded K 222 -> 224)
    {
        int n = 2 * G3H_ * KENC_;
        cvt_pad_kernel<<<blocks(n), 256, 0, stream>>>(W_ih0, Wih0f, 222, KENC_, n);
    }
    {
        int n = 2 * G3H_ * H_;
        cvt_kernel<<<blocks(n), 256, 0, stream>>>(W_hh0, Whh0f, n);
        cvt_kernel<<<blocks(n), 256, 0, stream>>>(W_ih1, Wih1f, n);
        cvt_kernel<<<blocks(n), 256, 0, stream>>>(W_hh1, Whh1f, n);
    }
    {
        int n = 2 * H_ * H_;
        cvt_kernel<<<blocks(n), 256, 0, stream>>>(W1, W1f, n);
        cvt_kernel<<<blocks(n), 256, 0, stream>>>(W2, W2f, n);
    }

    // sequential scan: one fused kernel per timestep, both groups in one grid
    for (int t = 0; t < T_; ++t) {
        step_kernel<<<BLOCKS0 + BLOCKS1, NTHREADS, 0, stream>>>(
            states, t,
            Wih0f, Whh0f, Wih1f, Whh1f, W1f, W2f,
            b_ih0, b_hh0, b_ih1, b_hh1, b1, b2, Wm, bm,
            h0g0, h1g0, h0g1, h1g1, accum);
    }

    final_kernel<<<1, 32, 0, stream>>>(accum, (float*)d_out);
}
